// BertWordPair_11433202941969
// MI455X (gfx1250) — matrix operand order repaired and checked by reading, not compiled, not run on hardware
//
#include <hip/hip_runtime.h>
#include <hip/hip_bf16.h>

// ---------------------------------------------------------------------------
// GAT (2-layer graph attention) for MI455X / gfx1250, wave32 + WMMA bf16.
// Big matmuls (x@W, attn@h) run on v_wmma_f32_16x16x32_bf16. The 4096x4096
// attention matrix is never materialized: softmax stats are computed per row,
// and the aggregation kernel rebuilds 16x32 P-tiles in LDS (bf16) on the fly.
// GEMM is software-pipelined (ping-pong fragment buffers) so WMMAs overlap
// in-flight global loads instead of draining loadcnt to zero each step.
// ---------------------------------------------------------------------------

#define GAT_N    4096
#define GAT_F    768
#define GAT_HID  768
#define GAT_NCLS 256
#define GAT_NH   3
#define GAT_CAT  (GAT_HID * GAT_NH)   // 2304

typedef __attribute__((ext_vector_type(16))) __bf16 v16bf;
typedef __attribute__((ext_vector_type(8)))  __bf16 v8bf;
typedef __attribute__((ext_vector_type(8)))  float  v8f;

union ABfrag { v16bf v; v8bf h[2]; };

__device__ __forceinline__ v8f wmma_bf16(const ABfrag& a, const ABfrag& b, v8f c) {
  return __builtin_amdgcn_wmma_f32_16x16x32_bf16(
      false, a.v, false, b.v, (short)0, c, false, false);
}

// ---------------------------------------------------------------------------
// fp32 -> bf16 (row-major, elementwise)
// ---------------------------------------------------------------------------
__global__ void cvt_f32_bf16(const float* __restrict__ src,
                             __bf16* __restrict__ dst, int n) {
  int i = blockIdx.x * blockDim.x + threadIdx.x;
  if (i < n) dst[i] = (__bf16)src[i];
}

// ---------------------------------------------------------------------------
// fp32 [K][N] -> bf16 transposed [N][K]  (LDS tiled transpose)
// ---------------------------------------------------------------------------
__global__ void transpose_cvt_bf16(const float* __restrict__ src,
                                   __bf16* __restrict__ dst, int K, int N) {
  __shared__ float tile[32][33];
  int n0 = blockIdx.x * 32, k0 = blockIdx.y * 32;
  for (int i = threadIdx.y; i < 32; i += 8) {
    int k = k0 + i, n = n0 + threadIdx.x;
    tile[i][threadIdx.x] = (k < K && n < N) ? src[(size_t)k * N + n] : 0.0f;
  }
  __syncthreads();
  for (int i = threadIdx.y; i < 32; i += 8) {
    int n = n0 + i, k = k0 + threadIdx.x;
    if (n < N && k < K) dst[(size_t)n * K + k] = (__bf16)tile[threadIdx.x][i];
  }
}

// ---------------------------------------------------------------------------
// WMMA bf16 GEMM:  C[M][Nfull] = A[M][K] @ Bt[N][K]^T
// One wave per 32x64 output tile (2 A-frags x 4 B-tiles = 8 WMMAs / k-step),
// double-buffered over two k-steps (requires K % 64 == 0).
// Writes fp32 C (optional) and bf16 transposed Ct[N][M] (optional).
// Fragment layouts per CDNA5 ISA 7.12.2 (wave32, 16-bit A 16x32 / B 32x16).
// ---------------------------------------------------------------------------
__global__ void __launch_bounds__(32)
wmma_gemm_bf16(const __bf16* __restrict__ A, const __bf16* __restrict__ Bt,
               int M, int K, int Nt64, int Nfull,
               float* __restrict__ C, __bf16* __restrict__ Ct) {
  int wid  = blockIdx.x;
  int nblk = wid % Nt64;
  int mblk = wid / Nt64;
  int lane = threadIdx.x & 31;
  int half = lane >> 4;
  int ln16 = lane & 15;
  int M0 = mblk * 32, N0 = nblk * 64;

  const __bf16* ap0 = A + (size_t)(M0 + ln16) * K + half * 8;       // rows M0..M0+15
  const __bf16* ap1 = ap0 + (size_t)16 * K;                         // rows M0+16..M0+31
  const __bf16* bp0 = Bt + (size_t)(N0 + ln16) * K + half * 8;

  v8f acc[2][4] = {};
  ABfrag a0[2], a1[2], b[2][4];

  auto loadA = [&](ABfrag& f, const __bf16* p, int k) {
    f.h[0] = *(const v8bf*)(p + k);        // K = k + half*8 + [0..7]
    f.h[1] = *(const v8bf*)(p + k + 16);   // K = k + 16 + half*8 + [0..7]
  };
  auto loadB = [&](ABfrag& f, int t, int k) {
    const __bf16* p = bp0 + (size_t)t * 16 * K + k;
    f.h[0] = *(const v8bf*)(p);            // K = k + half*8 + [0..7]
    f.h[1] = *(const v8bf*)(p + 16);       // K = k + 16 + half*8 + [0..7]
  };

  // prologue: buffer 0 holds k = 0
  loadA(a0[0], ap0, 0);
  loadA(a1[0], ap1, 0);
#pragma unroll
  for (int t = 0; t < 4; ++t) loadB(b[0][t], t, 0);

  for (int k = 0; k < K; k += 64) {
    // prefetch buffer 1 (k+32) while buffer 0 computes
    loadA(a0[1], ap0, k + 32);
    loadA(a1[1], ap1, k + 32);
#pragma unroll
    for (int t = 0; t < 4; ++t) loadB(b[1][t], t, k + 32);
#pragma unroll
    for (int t = 0; t < 4; ++t) {
      acc[0][t] = wmma_bf16(a0[0], b[0][t], acc[0][t]);
      acc[1][t] = wmma_bf16(a1[0], b[0][t], acc[1][t]);
    }
    // prefetch buffer 0 (k+64) while buffer 1 computes
    if (k + 64 < K) {
      loadA(a0[0], ap0, k + 64);
      loadA(a1[0], ap1, k + 64);
#pragma unroll
      for (int t = 0; t < 4; ++t) loadB(b[0][t], t, k + 64);
    }
#pragma unroll
    for (int t = 0; t < 4; ++t) {
      acc[0][t] = wmma_bf16(a0[1], b[1][t], acc[0][t]);
      acc[1][t] = wmma_bf16(a1[1], b[1][t], acc[1][t]);
    }
  }

#pragma unroll
  for (int i = 0; i < 2; ++i) {
#pragma unroll
    for (int t = 0; t < 4; ++t) {
#pragma unroll
      for (int r = 0; r < 8; ++r) {
        int row = M0 + i * 16 + half * 8 + r;
        int col = N0 + t * 16 + ln16;
        float v = acc[i][t][r];
        if (C)  C[(size_t)row * Nfull + col] = v;
        if (Ct) Ct[(size_t)col * M + row] = (__bf16)v;
      }
    }
  }
}

// ---------------------------------------------------------------------------
// Attention scores: s1[m] = h[m,:].a1 ; s2[m] = h[m,:].a2  (avec = [a1|a2])
// ---------------------------------------------------------------------------
__global__ void gat_scores(const float* __restrict__ h,
                           const float* __restrict__ avec, int D,
                           float* __restrict__ s1, float* __restrict__ s2) {
  int m = blockIdx.x, tid = threadIdx.x;
  const float* hr = h + (size_t)m * D;
  float p1 = 0.f, p2 = 0.f;
  for (int c = tid; c < D; c += blockDim.x) {
    float v = hr[c];
    p1 += v * avec[c];
    p2 += v * avec[D + c];
  }
  __shared__ float r1[256], r2[256];
  r1[tid] = p1; r2[tid] = p2;
  __syncthreads();
  for (int s = 128; s > 0; s >>= 1) {
    if (tid < s) { r1[tid] += r1[tid + s]; r2[tid] += r2[tid + s]; }
    __syncthreads();
  }
  if (tid == 0) { s1[m] = r1[0]; s2[m] = r2[0]; }
}

// ---------------------------------------------------------------------------
// Per-row masked-softmax stats (online max/sum; leaky_relu slope==1 => identity)
// ---------------------------------------------------------------------------
__global__ void gat_softmax_stats(const float* __restrict__ s1,
                                  const float* __restrict__ s2,
                                  const int* __restrict__ adj, int N,
                                  float* __restrict__ rmax,
                                  float* __restrict__ rinv) {
  int m = blockIdx.x, tid = threadIdx.x;
  const int* arow = adj + (size_t)m * N;
  float base = s1[m];
  float mx = -3.0e38f, sm = 0.f;
  for (int n = tid; n < N; n += blockDim.x) {
    if (arow[n] > 0) {
      float e = base + s2[n];
      if (e > mx) { sm *= __expf(mx - e); mx = e; }
      sm += __expf(e - mx);
    }
  }
  __shared__ float smx[256], ssm[256];
  smx[tid] = mx; ssm[tid] = sm;
  __syncthreads();
  for (int s = 128; s > 0; s >>= 1) {
    if (tid < s) {
      float m1 = smx[tid], m2 = smx[tid + s];
      float Mx = fmaxf(m1, m2);
      ssm[tid] = ssm[tid] * __expf(m1 - Mx) + ssm[tid + s] * __expf(m2 - Mx);
      smx[tid] = Mx;
    }
    __syncthreads();
  }
  if (tid == 0) {
    rmax[m] = smx[0];
    rinv[m] = ssm[0] > 0.f ? 1.f / ssm[0] : 0.f;
  }
}

// ---------------------------------------------------------------------------
// Fused attention aggregation (flash-style):  out[16, Nout] += P @ h
// Workgroup = 16 rows, 8 waves. Per 32-col step: B fragments (Ht, global) are
// issued FIRST, then the bf16 P-tile is rebuilt in LDS (exp/mask VALU work
// hides the global latency), then WMMAs consume both.
// NT=6 -> head layer (768 cols, ELU + bf16 concat store)
// NT=2 -> output layer (256 cols, raw fp32 store)
// ---------------------------------------------------------------------------
template <int NT>
__global__ void __launch_bounds__(256)
gat_aggregate(const float* __restrict__ s1, const float* __restrict__ s2,
              const int* __restrict__ adj,
              const float* __restrict__ rmax, const float* __restrict__ rinv,
              const __bf16* __restrict__ Ht, int Mtot, int Nadj,
              __bf16* dst_bf, int dst_stride, int dst_coloff,
              float* dst_f32, int f32_stride) {
  constexpr int PPAD = 40;              // 80B row stride: 16B aligned, spreads banks
  __shared__ __bf16 P[16][PPAD];
  int m0   = blockIdx.x * 16;
  int tid  = threadIdx.x;
  int wave = tid >> 5, lane = tid & 31;
  int half = lane >> 4, ln16 = lane & 15;

  // Hoisted B-fragment base pointers (per tile this wave owns).
  const __bf16* bbase[NT];
#pragma unroll
  for (int t = 0; t < NT; ++t)
    bbase[t] = Ht + (size_t)((wave * NT + t) * 16 + ln16) * Mtot + half * 8;

  // Each thread fills 2 fixed P elements per step: hoist row constants.
  int pm0 = tid >> 5,          kk0 = tid & 31;
  int pm1 = (tid + 256) >> 5,  kk1 = (tid + 256) & 31;
  int row0 = m0 + pm0, row1 = m0 + pm1;
  float sh0 = s1[row0] - rmax[row0], ri0 = rinv[row0];
  float sh1 = s1[row1] - rmax[row1], ri1 = rinv[row1];
  const int* arow0 = adj + (size_t)row0 * Nadj;
  const int* arow1 = adj + (size_t)row1 * Nadj;

  v8f acc[NT] = {};
  for (int nb = 0; nb < Nadj; nb += 32) {
    // 1) issue B-fragment global loads first (no LDS dependency)
    ABfrag b[NT];
#pragma unroll
    for (int t = 0; t < NT; ++t) {
      const __bf16* bp = bbase[t] + nb;
      b[t].h[0] = *(const v8bf*)(bp);
      b[t].h[1] = *(const v8bf*)(bp + 16);
    }

    // 2) rebuild P tile (exp/mask) -> LDS; hides the load latency
    float v0 = 0.f, v1 = 0.f;
    if (arow0[nb + kk0] > 0) v0 = __expf(sh0 + s2[nb + kk0]) * ri0;
    if (arow1[nb + kk1] > 0) v1 = __expf(sh1 + s2[nb + kk1]) * ri1;
    P[pm0][kk0] = (__bf16)v0;
    P[pm1][kk1] = (__bf16)v1;
    __syncthreads();

    // 3) A fragment from LDS, then the WMMAs
    ABfrag a;
    a.h[0] = *(const v8bf*)&P[ln16][half * 8];
    a.h[1] = *(const v8bf*)&P[ln16][16 + half * 8];
#pragma unroll
    for (int t = 0; t < NT; ++t)
      acc[t] = wmma_bf16(a, b[t], acc[t]);
    __syncthreads();
  }

#pragma unroll
  for (int t = 0; t < NT; ++t) {
#pragma unroll
    for (int r = 0; r < 8; ++r) {
      int row = m0 + half * 8 + r;
      int col = (wave * NT + t) * 16 + ln16;
      float v = acc[t][r];
      if (dst_bf) {   // head layer: ELU then bf16 concat store
        float e = v > 0.f ? v : (__expf(v) - 1.f);
        dst_bf[(size_t)row * dst_stride + dst_coloff + col] = (__bf16)e;
      }
      if (dst_f32)    // output layer: raw fp32
        dst_f32[(size_t)row * f32_stride + col] = v;
    }
  }
}

// ---------------------------------------------------------------------------
// Final: out = log_softmax(elu(hprime), axis=-1), C == blockDim == 256
// ---------------------------------------------------------------------------
__global__ void elu_log_softmax(const float* __restrict__ in,
                                float* __restrict__ out, int C) {
  int m = blockIdx.x, tid = threadIdx.x;
  float v = in[(size_t)m * C + tid];
  v = v > 0.f ? v : (__expf(v) - 1.f);
  __shared__ float red[256];
  red[tid] = v;
  __syncthreads();
  for (int s = 128; s > 0; s >>= 1) {
    if (tid < s) red[tid] = fmaxf(red[tid], red[tid + s]);
    __syncthreads();
  }
  float mx = red[0];
  __syncthreads();
  red[tid] = __expf(v - mx);
  __syncthreads();
  for (int s = 128; s > 0; s >>= 1) {
    if (tid < s) red[tid] += red[tid + s];
    __syncthreads();
  }
  float lse = mx + __logf(red[0]);
  out[(size_t)m * C + tid] = v - lse;
}

// ---------------------------------------------------------------------------
// Host orchestration
// ---------------------------------------------------------------------------
extern "C" void kernel_launch(void* const* d_in, const int* in_sizes, int n_in,
                              void* d_out, int out_size, void* d_ws, size_t ws_size,
                              hipStream_t stream) {
  (void)in_sizes; (void)n_in; (void)out_size; (void)ws_size;
  const int N = GAT_N, F = GAT_F, HID = GAT_HID, NCLS = GAT_NCLS, CAT = GAT_CAT;

  const float* x   = (const float*)d_in[0];
  const int*   adj = (const int*)  d_in[1];
  const float* Wh  = (const float*)d_in[2];   // (NH, 1, F, HID)
  const float* ah  = (const float*)d_in[3];   // (NH, 2*HID, 1)
  const float* Wo  = (const float*)d_in[4];   // (1, CAT, NCLS)
  const float* ao  = (const float*)d_in[5];   // (2*NCLS, 1)

  char* ws = (char*)d_ws;
  auto alloc = [&](size_t bytes) -> char* {
    char* p = ws; ws += (bytes + 255) & ~(size_t)255; return p;
  };
  __bf16* x_bf   = (__bf16*)alloc((size_t)N * F * 2);
  __bf16* Wt_h   = (__bf16*)alloc((size_t)GAT_NH * F * HID * 2);   // [h][HID][F]
  __bf16* Wt_o   = (__bf16*)alloc((size_t)NCLS * CAT * 2);         // [NCLS][CAT]
  float*  h_f    = (float*) alloc((size_t)GAT_NH * N * HID * 4);   // fp32 h per head
  __bf16* ht_bf  = (__bf16*)alloc((size_t)GAT_NH * HID * N * 2);   // transposed bf16
  float*  s1     = (float*) alloc((size_t)GAT_NH * N * 4);
  float*  s2     = (float*) alloc((size_t)GAT_NH * N * 4);
  float*  rmx    = (float*) alloc((size_t)GAT_NH * N * 4);
  float*  rin    = (float*) alloc((size_t)GAT_NH * N * 4);
  __bf16* xcat   = (__bf16*)alloc((size_t)N * CAT * 2);            // elu(h'), concat
  float*  hout   = (float*) alloc((size_t)N * NCLS * 4);
  __bf16* htout  = (__bf16*)alloc((size_t)NCLS * N * 2);
  float*  s1o    = (float*) alloc((size_t)N * 4);
  float*  s2o    = (float*) alloc((size_t)N * 4);
  float*  rmxo   = (float*) alloc((size_t)N * 4);
  float*  rino   = (float*) alloc((size_t)N * 4);
  float*  hpo    = (float*) alloc((size_t)N * NCLS * 4);

  // 1) precision conversion
  cvt_f32_bf16<<<(N * F + 255) / 256, 256, 0, stream>>>(x, x_bf, N * F);
  for (int h = 0; h < GAT_NH; ++h)
    transpose_cvt_bf16<<<dim3(HID / 32, F / 32), dim3(32, 8), 0, stream>>>(
        Wh + (size_t)h * F * HID, Wt_h + (size_t)h * HID * F, F, HID);
  transpose_cvt_bf16<<<dim3(NCLS / 32, CAT / 32), dim3(32, 8), 0, stream>>>(
      Wo, Wt_o, CAT, NCLS);

  // 2) per-head: h = x@W, scores, softmax stats, fused attn@h (+ELU -> xcat)
  for (int h = 0; h < GAT_NH; ++h) {
    wmma_gemm_bf16<<<(N / 32) * (HID / 64), 32, 0, stream>>>(
        x_bf, Wt_h + (size_t)h * HID * F, N, F, HID / 64, HID,
        h_f + (size_t)h * N * HID, ht_bf + (size_t)h * HID * N);
    gat_scores<<<N, 256, 0, stream>>>(
        h_f + (size_t)h * N * HID, ah + (size_t)h * 2 * HID, HID,
        s1 + (size_t)h * N, s2 + (size_t)h * N);
    gat_softmax_stats<<<N, 256, 0, stream>>>(
        s1 + (size_t)h * N, s2 + (size_t)h * N, adj, N,
        rmx + (size_t)h * N, rin + (size_t)h * N);
    gat_aggregate<6><<<N / 16, 256, 0, stream>>>(
        s1 + (size_t)h * N, s2 + (size_t)h * N, adj,
        rmx + (size_t)h * N, rin + (size_t)h * N,
        ht_bf + (size_t)h * HID * N, N, N,
        xcat, CAT, h * HID, (float*)nullptr, 0);
  }

  // 3) output layer
  wmma_gemm_bf16<<<(N / 32) * (NCLS / 64), 32, 0, stream>>>(
      xcat, Wt_o, N, CAT, NCLS / 64, NCLS, hout, htout);
  gat_scores<<<N, 256, 0, stream>>>(hout, ao, NCLS, s1o, s2o);
  gat_softmax_stats<<<N, 256, 0, stream>>>(s1o, s2o, adj, N, rmxo, rino);
  gat_aggregate<2><<<N / 16, 256, 0, stream>>>(
      s1o, s2o, adj, rmxo, rino, htout, N, N,
      (__bf16*)nullptr, 0, 0, hpo, NCLS);

  // 4) elu + log_softmax -> d_out
  elu_log_softmax<<<N, 256, 0, stream>>>(hpo, (float*)d_out, NCLS);
}
